// StreamingExpert_12352325943572
// MI455X (gfx1250) — matrix operand order repaired and checked by reading, not compiled, not run on hardware
//
#include <hip/hip_runtime.h>
#include <hip/hip_bf16.h>
#include <stdint.h>

// CDNA5 (gfx1250) WMMA fragment types (wave32)
typedef __attribute__((ext_vector_type(16))) __bf16 v16bf;
typedef __attribute__((ext_vector_type(8)))  float  v8f;
typedef __attribute__((ext_vector_type(4)))  int    v4i;

union Frag16 { v16bf v; uint4 u[2]; };

__device__ __forceinline__ unsigned short f2bf(float f) {
  unsigned int u = __float_as_uint(f);
  u += 0x7FFFu + ((u >> 16) & 1u);
  return (unsigned short)(u >> 16);
}

// ---------------------------------------------------------------------------
// gfx1250 async global->LDS copy (ASYNCcnt-tracked DMA, no VGPR round trip).
// Builtin prototype (from compiler diagnostic): (v4i AS1*, v4i AS3*, Imm, Imm).
// ---------------------------------------------------------------------------
typedef __attribute__((address_space(1))) v4i* as1_v4i;
typedef __attribute__((address_space(3))) v4i* as3_v4i;

__device__ __forceinline__ void async_copy16(const void* g, void* l) {
#if defined(__gfx1250__) && __has_builtin(__builtin_amdgcn_global_load_async_to_lds_b128)
  __builtin_amdgcn_global_load_async_to_lds_b128(
      (as1_v4i)(uintptr_t)g, (as3_v4i)(uintptr_t)l, 0, 0);
#else
  *(uint4*)l = *(const uint4*)g;
#endif
}

template <int N>
__device__ __forceinline__ void wait_async() {
#if defined(__gfx1250__) && __has_builtin(__builtin_amdgcn_s_wait_asynccnt)
  __builtin_amdgcn_s_wait_asynccnt(N);
#endif
}

// ---------------------------------------------------------------------------
// Kernel 1: hidden_states f32 -> bf16, vectorized x4
// ---------------------------------------------------------------------------
__global__ __launch_bounds__(256) void cvt_f32_to_bf16_kernel(
    const float* __restrict__ src, unsigned short* __restrict__ dst, long long n4) {
  long long i = (long long)blockIdx.x * 256 + threadIdx.x;
  if (i >= n4) return;
  float4 f = ((const float4*)src)[i];
  ushort4 o;
  o.x = f2bf(f.x); o.y = f2bf(f.y); o.z = f2bf(f.z); o.w = f2bf(f.w);
  ((ushort4*)dst)[i] = o;
}

// ---------------------------------------------------------------------------
// Kernel 2: group-asym dequant  w = (q - zp) * scale  -> bf16, vectorized x4
// ---------------------------------------------------------------------------
__global__ __launch_bounds__(256) void dequant4_kernel(
    const int* __restrict__ q, const float* __restrict__ scale,
    const float* __restrict__ zp, unsigned short* __restrict__ dst,
    int in_dim, long long n4) {
  long long i = (long long)blockIdx.x * 256 + threadIdx.x;
  if (i >= n4) return;
  long long base = i * 4;
  long long row  = base / in_dim;
  int col = (int)(base - row * (long long)in_dim);
  long long g = row * (long long)(in_dim >> 7) + (col >> 7);
  float s = scale[g];
  float z = zp[g];
  int4 qv = ((const int4*)q)[i];
  ushort4 o;
  o.x = f2bf(((float)qv.x - z) * s);
  o.y = f2bf(((float)qv.y - z) * s);
  o.z = f2bf(((float)qv.z - z) * s);
  o.w = f2bf(((float)qv.w - z) * s);
  ((ushort4*)dst)[i] = o;
}

// ---------------------------------------------------------------------------
// WMMA GEMM tiling: block = 256 threads = 8 wave32, tile 64(M) x 128(N) x 64(K)
// wave grid 2x4, each wave: 32x32 via 2x2 wmma_f32_16x16x32_bf16 per operand.
// Double-buffered LDS, rows padded to 72 shorts (144B) against bank conflicts.
// ---------------------------------------------------------------------------
#define BM  64
#define BN  128
#define BK  64
#define LDW 72

__device__ __forceinline__ v16bf load_a_frag(const unsigned short* sA, int row,
                                             int kk, int hi) {
  Frag16 f;
  const unsigned short* p = sA + row * LDW + kk + hi * 16;
  f.u[0] = *(const uint4*)(p);
  f.u[1] = *(const uint4*)(p + 8);
  return f.v;
}

__device__ __forceinline__ v16bf load_b_frag(const unsigned short* sB, int nrow,
                                             int kk, int hi) {
  Frag16 f;
  const unsigned short* p = sB + nrow * LDW + kk + hi * 16;
  f.u[0] = *(const uint4*)(p);
  f.u[1] = *(const uint4*)(p + 8);
  return f.v;
}

// ---------------------------------------------------------------------------
// Kernel 3: fused gate/up GEMMs + SwiGLU epilogue.
//   ACT[T,I] = silu(X @ Wg^T) * (X @ Wu^T), bf16
// Per-iteration async loads per thread: A:2 + G:4 + U:4 = 10.
// ---------------------------------------------------------------------------
__global__ __launch_bounds__(256) void gateup_wmma_kernel(
    const unsigned short* __restrict__ X,   // [T,H] bf16
    const unsigned short* __restrict__ Wg,  // [I,H] bf16
    const unsigned short* __restrict__ Wu,  // [I,H] bf16
    unsigned short* __restrict__ ACT,       // [T,I] bf16
    int Hdim, int Idim) {
  __shared__ __align__(16) unsigned short sA[2][BM * LDW];
  __shared__ __align__(16) unsigned short sG[2][BN * LDW];
  __shared__ __align__(16) unsigned short sU[2][BN * LDW];

  const int tid  = threadIdx.x;
  const int lane = tid & 31;
  const int wid  = tid >> 5;
  const int wm   = wid >> 2;
  const int wn   = wid & 3;
  const int l15  = lane & 15;
  const int hi   = lane >> 4;

  const int bm = blockIdx.x * BM;
  const int bn = blockIdx.y * BN;

  auto issue = [&](int buf, int kc) {
#pragma unroll
    for (int it = 0; it < 2; ++it) {           // A tile: 64 rows x 128B
      int flat = tid + it * 256;
      int r = flat >> 3, c = flat & 7;
      async_copy16(X + (long long)(bm + r) * Hdim + kc + c * 8,
                   &sA[buf][r * LDW + c * 8]);
    }
#pragma unroll
    for (int it = 0; it < 4; ++it) {           // G,U tiles: 128 rows x 128B
      int flat = tid + it * 256;
      int r = flat >> 3, c = flat & 7;
      long long go = (long long)(bn + r) * Hdim + kc + c * 8;
      async_copy16(Wg + go, &sG[buf][r * LDW + c * 8]);
      async_copy16(Wu + go, &sU[buf][r * LDW + c * 8]);
    }
  };

  v8f accg[2][2], accu[2][2];
  const v8f vzero = {0.f, 0.f, 0.f, 0.f, 0.f, 0.f, 0.f, 0.f};
#pragma unroll
  for (int mi = 0; mi < 2; ++mi)
#pragma unroll
    for (int ni = 0; ni < 2; ++ni) { accg[mi][ni] = vzero; accu[mi][ni] = vzero; }

  const int niter = Hdim / BK;
  issue(0, 0);
  issue(1, BK);

  for (int i = 0; i < niter; ++i) {
    const int cur = i & 1;
    // oldest chunk complete: leave the (optional) next chunk's 10 loads in flight
    if (i + 1 < niter) wait_async<10>(); else wait_async<0>();
    __syncthreads();

    const unsigned short* cA = sA[cur];
    const unsigned short* cG = sG[cur];
    const unsigned short* cU = sU[cur];
#pragma unroll
    for (int kk = 0; kk < BK; kk += 32) {
      v16bf a[2], bg[2], bu[2];
#pragma unroll
      for (int mi = 0; mi < 2; ++mi)
        a[mi] = load_a_frag(cA, wm * 32 + mi * 16 + l15, kk, hi);
#pragma unroll
      for (int ni = 0; ni < 2; ++ni) {
        bg[ni] = load_b_frag(cG, wn * 32 + ni * 16 + l15, kk, hi);
        bu[ni] = load_b_frag(cU, wn * 32 + ni * 16 + l15, kk, hi);
      }
#pragma unroll
      for (int mi = 0; mi < 2; ++mi)
#pragma unroll
        for (int ni = 0; ni < 2; ++ni) {
          accg[mi][ni] = __builtin_amdgcn_wmma_f32_16x16x32_bf16(
              false, a[mi], false, bg[ni], (short)0, accg[mi][ni], false, false);
          accu[mi][ni] = __builtin_amdgcn_wmma_f32_16x16x32_bf16(
              false, a[mi], false, bu[ni], (short)0, accu[mi][ni], false, false);
        }
    }
    __syncthreads();   // all waves done reading buf[cur]
    if (i + 2 < niter) issue(cur, (i + 2) * BK);
  }

#pragma unroll
  for (int mi = 0; mi < 2; ++mi)
#pragma unroll
    for (int ni = 0; ni < 2; ++ni)
#pragma unroll
      for (int r = 0; r < 8; ++r) {
        int M = bm + wm * 32 + mi * 16 + hi * 8 + r;
        int N = bn + wn * 32 + ni * 16 + l15;
        float g = accg[mi][ni][r];
        float u = accu[mi][ni][r];
        float sg = g / (1.f + __expf(-g));
        ACT[(long long)M * Idim + N] = f2bf(sg * u);
      }
}

// ---------------------------------------------------------------------------
// Kernel 4: down GEMM: out[T,H] f32 = ACT[T,I] @ Wd[H,I]^T
// Per-iteration async loads per thread: A:2 + B:4 = 6.
// ---------------------------------------------------------------------------
__global__ __launch_bounds__(256) void down_wmma_kernel(
    const unsigned short* __restrict__ A,   // [T,I] bf16
    const unsigned short* __restrict__ Wd,  // [H,I] bf16
    float* __restrict__ out,                // [T,H] f32
    int Idim, int Hdim) {
  __shared__ __align__(16) unsigned short sA[2][BM * LDW];
  __shared__ __align__(16) unsigned short sB[2][BN * LDW];

  const int tid  = threadIdx.x;
  const int lane = tid & 31;
  const int wid  = tid >> 5;
  const int wm   = wid >> 2;
  const int wn   = wid & 3;
  const int l15  = lane & 15;
  const int hi   = lane >> 4;

  const int bm = blockIdx.x * BM;
  const int bn = blockIdx.y * BN;

  auto issue = [&](int buf, int kc) {
#pragma unroll
    for (int it = 0; it < 2; ++it) {
      int flat = tid + it * 256;
      int r = flat >> 3, c = flat & 7;
      async_copy16(A + (long long)(bm + r) * Idim + kc + c * 8,
                   &sA[buf][r * LDW + c * 8]);
    }
#pragma unroll
    for (int it = 0; it < 4; ++it) {
      int flat = tid + it * 256;
      int r = flat >> 3, c = flat & 7;
      async_copy16(Wd + (long long)(bn + r) * Idim + kc + c * 8,
                   &sB[buf][r * LDW + c * 8]);
    }
  };

  v8f acc[2][2];
  const v8f vzero = {0.f, 0.f, 0.f, 0.f, 0.f, 0.f, 0.f, 0.f};
#pragma unroll
  for (int mi = 0; mi < 2; ++mi)
#pragma unroll
    for (int ni = 0; ni < 2; ++ni) acc[mi][ni] = vzero;

  const int niter = Idim / BK;
  issue(0, 0);
  issue(1, BK);

  for (int i = 0; i < niter; ++i) {
    const int cur = i & 1;
    if (i + 1 < niter) wait_async<6>(); else wait_async<0>();
    __syncthreads();

    const unsigned short* cA = sA[cur];
    const unsigned short* cB = sB[cur];
#pragma unroll
    for (int kk = 0; kk < BK; kk += 32) {
      v16bf a[2], b[2];
#pragma unroll
      for (int mi = 0; mi < 2; ++mi)
        a[mi] = load_a_frag(cA, wm * 32 + mi * 16 + l15, kk, hi);
#pragma unroll
      for (int ni = 0; ni < 2; ++ni)
        b[ni] = load_b_frag(cB, wn * 32 + ni * 16 + l15, kk, hi);
#pragma unroll
      for (int mi = 0; mi < 2; ++mi)
#pragma unroll
        for (int ni = 0; ni < 2; ++ni)
          acc[mi][ni] = __builtin_amdgcn_wmma_f32_16x16x32_bf16(
              false, a[mi], false, b[ni], (short)0, acc[mi][ni], false, false);
    }
    __syncthreads();
    if (i + 2 < niter) issue(cur, (i + 2) * BK);
  }

#pragma unroll
  for (int mi = 0; mi < 2; ++mi)
#pragma unroll
    for (int ni = 0; ni < 2; ++ni)
#pragma unroll
      for (int r = 0; r < 8; ++r) {
        int M = bm + wm * 32 + mi * 16 + hi * 8 + r;
        int N = bn + wn * 32 + ni * 16 + l15;
        out[(long long)M * Hdim + N] = acc[mi][ni][r];
      }
}

// ---------------------------------------------------------------------------
extern "C" void kernel_launch(void* const* d_in, const int* in_sizes, int n_in,
                              void* d_out, int out_size, void* d_ws, size_t ws_size,
                              hipStream_t stream) {
  (void)in_sizes; (void)n_in; (void)out_size; (void)ws_size;

  constexpr long long T = 8192, H = 2048, I = 5632;

  const float* hidden = (const float*)d_in[0];
  const int*   gate_q = (const int*)d_in[1];
  const float* gate_s = (const float*)d_in[2];
  const float* gate_z = (const float*)d_in[3];
  const int*   up_q   = (const int*)d_in[4];
  const float* up_s   = (const float*)d_in[5];
  const float* up_z   = (const float*)d_in[6];
  const int*   down_q = (const int*)d_in[7];
  const float* down_s = (const float*)d_in[8];
  const float* down_z = (const float*)d_in[9];
  float* out = (float*)d_out;

  // workspace: X16 | Wg | Wu | Wd | ACT  (~186 MB; weights 69MB -> L2-resident)
  char* ws = (char*)d_ws;
  size_t off = 0;
  unsigned short* X16 = (unsigned short*)(ws + off); off += (size_t)T * H * 2;
  unsigned short* WgB = (unsigned short*)(ws + off); off += (size_t)I * H * 2;
  unsigned short* WuB = (unsigned short*)(ws + off); off += (size_t)I * H * 2;
  unsigned short* WdB = (unsigned short*)(ws + off); off += (size_t)H * I * 2;
  unsigned short* ACT = (unsigned short*)(ws + off); off += (size_t)T * I * 2;

  {
    long long n4 = (T * H) / 4;
    cvt_f32_to_bf16_kernel<<<(unsigned)((n4 + 255) / 256), 256, 0, stream>>>(hidden, X16, n4);
  }
  {
    long long n4 = (I * H) / 4;
    unsigned nb = (unsigned)((n4 + 255) / 256);
    dequant4_kernel<<<nb, 256, 0, stream>>>(gate_q, gate_s, gate_z, WgB, (int)H, n4);
    dequant4_kernel<<<nb, 256, 0, stream>>>(up_q,   up_s,   up_z,   WuB, (int)H, n4);
    dequant4_kernel<<<nb, 256, 0, stream>>>(down_q, down_s, down_z, WdB, (int)I, n4);
  }
  gateup_wmma_kernel<<<dim3((unsigned)(T / BM), (unsigned)(I / BN)), 256, 0, stream>>>(
      X16, WgB, WuB, ACT, (int)H, (int)I);
  down_wmma_kernel<<<dim3((unsigned)(T / BM), (unsigned)(H / BN)), 256, 0, stream>>>(
      ACT, WdB, out, (int)I, (int)H);
}
